// ModelClass_14181982011739
// MI455X (gfx1250) — matrix operand order repaired and checked by reading, not compiled, not run on hardware
//
#include <hip/hip_runtime.h>

typedef unsigned short u16;
typedef __attribute__((ext_vector_type(16))) __bf16 v16bf;
typedef __attribute__((ext_vector_type(8)))  float  v8f;

#define AINLINE __attribute__((always_inline)) __device__ inline

#define NTHREADS 256
#define NWAVES   8
#define GB       8            // graphs per workgroup
#define N0       30           // max nodes per graph
#define FMAX     20           // feature stride (>= 18)
#define RMAX     (GB*N0)      // 240 node-rows max
#define BGRAPH   131072
#define LRELU_NEG 0.01f
#define BN_EPS    1e-5f
#define HID       40
#define NEG_INF  (-3.402823466e38f)
#define POS_INF  ( 3.402823466e38f)

// ---------------- parameter table (canonical order) ----------------
// JAX pytree (sorted keys) leaf order inside 'params':
//   last_disc tsum @0 (28 leaves), level0 @28 (56), level1 @84 (56)
// tsum base T:  cnu c @ T+12c:  emb{b1,b2,w1,w2}=+0..3  glob=+4..7  out=+8..11
//               disc{b1,b2,w1,w2} = T+24..27
// level base L: disc tsum @L ; gin0 eps=L+28, nn(bn)=L+29..36
//               gin1 eps=L+37, nn=L+38..45 ; out(bn)=L+46..53 ; pool b=L+54 w=L+55
// bn-ffn leaf order: b1,b2,bt1,g1,m1,v1,w1,w2
struct KParams {
  const float* x;        // [B,30,3]
  const float* cond;     // [B,5]
  const float* p[140];
};

// ---------------- address-space hints ----------------
AINLINE void assume_lds(const void* p) {
  __builtin_assume(__builtin_amdgcn_is_shared(p));
}
AINLINE void assume_glb(const void* p) {
  __builtin_assume(!__builtin_amdgcn_is_shared(p));
  __builtin_assume(!__builtin_amdgcn_is_private(p));
}

// ---------------- small helpers ----------------
AINLINE u16 f2bf(float f) {
  unsigned u = __float_as_uint(f);
  unsigned r = u + 0x7FFFu + ((u >> 16) & 1u);   // RNE
  return (u16)(r >> 16);
}
AINLINE float bf2f(u16 h) {
  return __uint_as_float(((unsigned)h) << 16);
}
AINLINE float lrelu(float v) { return v > 0.f ? v : LRELU_NEG * v; }

// B-fragment (K x 16) for v_wmma_f32_16x16x32_bf16, built from global f32 weights.
// Layout: lanes 0-15 -> K = k0+e (e<8), k0+16+(e-8) (e>=8), N = n0+lane ; lanes 16-31 -> same +8.
AINLINE v16bf load_wfrag(const float* W, int k0, int n0,
                         int K, int N, int ld, int lane) {
  assume_glb(W);
  int n  = n0 + (lane & 15);
  int kb = k0 + ((lane >> 4) << 3);
  v16bf f;
#pragma unroll
  for (int e = 0; e < 16; ++e) {
    int k = kb + (e < 8 ? e : e + 8);
    float v = (k < K && n < N) ? W[k * ld + n] : 0.f;
    union { u16 s; __bf16 b; } cv; cv.s = f2bf(v);
    f[e] = cv.b;
  }
  return f;
}

// A-fragment from LDS: p already points at row start + (lane>=16 ? 8 : 0).
// elements e<8 -> p[e] (K=kb..kb+7), e>=8 -> p[16+e-8] (K=kb+16..kb+23)
AINLINE v16bf load_afrag(const u16* p) {
  assume_lds(p);
  v16bf f;
#pragma unroll
  for (int e = 0; e < 8; ++e) {
    union { u16 s; __bf16 b; } a, b2;
    a.s  = p[e];
    b2.s = p[16 + e];
    f[e]     = a.b;
    f[e + 8] = b2.b;
  }
  return f;
}

struct FfnW {
  const float *w1, *b1, *w2, *b2;
  const float *g1, *bt1, *m1, *v1;   // batchnorm (null = none)
  int nin, nout;
};
AINLINE FfnW ffnP(const KParams& kp, int b, int nin, int nout) {
  FfnW w; w.w1 = kp.p[b+2]; w.b1 = kp.p[b+0]; w.w2 = kp.p[b+3]; w.b2 = kp.p[b+1];
  w.g1 = nullptr; w.bt1 = nullptr; w.m1 = nullptr; w.v1 = nullptr;
  w.nin = nin; w.nout = nout; return w;
}
AINLINE FfnW ffnBN(const KParams& kp, int b, int nin, int nout) {
  FfnW w; w.w1 = kp.p[b+6]; w.b1 = kp.p[b+0]; w.w2 = kp.p[b+7]; w.b2 = kp.p[b+1];
  w.g1 = kp.p[b+3]; w.bt1 = kp.p[b+2]; w.m1 = kp.p[b+4]; w.v1 = kp.p[b+5];
  w.nin = nin; w.nout = nout; return w;
}

// scalar 2-layer FFN (per-graph tiny paths: glob 4->40->5, disc f->40->1)
AINLINE void ffn_scalar(const float* in, FfnW W, bool leakyOut, float* outv) {
  assume_glb(W.w1); assume_glb(W.b1); assume_glb(W.w2); assume_glb(W.b2);
  float h[HID];
  for (int j = 0; j < HID; ++j) {
    float v = W.b1[j];
    for (int i = 0; i < W.nin; ++i) v += in[i] * W.w1[i * HID + j];
    h[j] = lrelu(v);
  }
  for (int o = 0; o < W.nout; ++o) {
    float v = W.b2[o];
    for (int j = 0; j < HID; ++j) v += h[j] * W.w2[j * W.nout + o];
    outv[o] = leakyOut ? lrelu(v) : v;
  }
}

// ---------------- WMMA 2-layer FFN over node-row tiles ----------------
// A: LDS [rowsPad][32] bf16 (inputs, K-padded with zeros). Hidden=40 (pad 48,
// cols 48..63 pre-zeroed). Each wave processes whole M-tiles independently
// (same-wave LDS ops are in-order, so no barrier between stage1/stage2).
// dstB: bf16 feature buffer addressed (g*N0+i)*FMAX+col with g=row/nPer.
// resMode: 0 none, 1 out+=res, 2 out+=2*res (tsum's x + central_update(x)).
AINLINE void ffn_wmma(FfnW W, const u16 (*A)[32], int rowsPad, int rowsValid,
                      u16* Hw, bool leakyOut,
                      u16* dstB, int nPer, const u16* resB, int resMode,
                      float* dstF, int ldF, int lane, int wave)
{
  assume_lds(A); assume_lds(Hw);
  assume_glb(W.w1); assume_glb(W.b1); assume_glb(W.w2); assume_glb(W.b2);
  if (W.g1) { assume_glb(W.g1); assume_glb(W.bt1); assume_glb(W.m1); assume_glb(W.v1); }
  if (dstB) assume_lds(dstB);
  if (resB) assume_lds(resB);
  if (dstF) assume_lds(dstF);

  const int hiOff = (lane >> 4) << 3;   // +8 bf16 elements for lanes 16..31
  const int mbase = (lane >> 4) << 3;   // C/D fragment row base
  const int ncol  = lane & 15;

  // weight fragments (once per call per wave)
  v16bf W1f[3];
#pragma unroll
  for (int t = 0; t < 3; ++t) W1f[t] = load_wfrag(W.w1, 0, 16 * t, W.nin, HID, HID, lane);
  const int nt2 = (W.nout + 15) >> 4;
  v16bf W2f[2][2];
  for (int kt = 0; kt < 2; ++kt)
    for (int nt = 0; nt < nt2; ++nt)
      W2f[kt][nt] = load_wfrag(W.w2, kt * 32, nt * 16, HID, W.nout, W.nout, lane);

  // per-lane loop-invariant postprocess constants:
  // hidden: hv = lrelu(c*hscale + hshift)  (affine covers bias + batchnorm;
  //         pad cols get scale=shift=0 -> store 0, keeps Hbuf zero-padding)
  float hscale[3], hshift[3];
#pragma unroll
  for (int t = 0; t < 3; ++t) {
    int col = t * 16 + ncol;
    float sc = 0.f, sh = 0.f;
    if (col < HID) {
      float b = W.b1[col];
      if (W.g1) {
        sc = rsqrtf(W.v1[col] + BN_EPS) * W.g1[col];
        sh = (b - W.m1[col]) * sc + W.bt1[col];
      } else { sc = 1.f; sh = b; }
    }
    hscale[t] = sc; hshift[t] = sh;
  }
  float oshift[2];
  bool  colOK[2];
  for (int nt = 0; nt < nt2; ++nt) {
    int col = nt * 16 + ncol;
    colOK[nt]  = (col < W.nout);
    oshift[nt] = colOK[nt] ? W.b2[col] : 0.f;
  }

  for (int mt = wave; mt < (rowsPad >> 4); mt += NWAVES) {
    int arow = mt * 16 + (lane & 15);
    v16bf a = load_afrag(&A[arow][0] + hiOff);
    // stage 1: A[16x32] @ W1 -> hidden 40 -> Hw bf16 [16][64]
#pragma unroll
    for (int t = 0; t < 3; ++t) {
      v8f c = {};
      c = __builtin_amdgcn_wmma_f32_16x16x32_bf16(false, a, false, W1f[t],
                                                  (short)0, c, false, false);
      int col = t * 16 + ncol;
#pragma unroll
      for (int r = 0; r < 8; ++r) {
        float hv = lrelu(fmaf(c[r], hscale[t], hshift[t]));
        Hw[(mbase + r) * 64 + col] = f2bf(hv);
      }
    }
    // stage 2: hidden[16x64] @ W2 (K = 2x32)
    for (int nt = 0; nt < nt2; ++nt) {
      v8f c = {};
#pragma unroll
      for (int kt = 0; kt < 2; ++kt) {
        v16bf ah = load_afrag(Hw + (lane & 15) * 64 + kt * 32 + hiOff);
        c = __builtin_amdgcn_wmma_f32_16x16x32_bf16(false, ah, false, W2f[kt][nt],
                                                    (short)0, c, false, false);
      }
      int col = nt * 16 + ncol;
#pragma unroll
      for (int r = 0; r < 8; ++r) {
        int grow = mt * 16 + mbase + r;
        if (grow < rowsValid && colOK[nt]) {
          float ov = c[r] + oshift[nt];
          if (leakyOut) ov = lrelu(ov);
          if (dstF) {
            dstF[grow * ldF + col] = ov;
          } else {
            int g  = grow / nPer, i2 = grow - g * nPer;
            int ad = (g * N0 + i2) * FMAX + col;
            if      (resMode == 1) ov += bf2f(resB[ad]);
            else if (resMode == 2) ov += 2.f * bf2f(resB[ad]);
            dstB[ad] = f2bf(ov);
          }
        }
      }
    }
  }
}

// ---------------- staging ----------------
AINLINE void stage_plain(const u16* buf, int n, int F, u16 (*Ab)[32],
                         int rowsPad, int tid) {
  assume_lds(buf); assume_lds(Ab);
  for (int r = tid; r < rowsPad; r += NTHREADS) {
    if (r < GB * n) {
      int g = r / n, i = r - g * n;
      const u16* src = buf + (g * N0 + i) * FMAX;
      for (int f = 0; f < F; ++f) Ab[r][f] = src[f];
      for (int c = F; c < 32; ++c) Ab[r][c] = 0;
    } else {
      for (int c = 0; c < 32; ++c) Ab[r][c] = 0;
    }
  }
}

AINLINE void stage_gin(const u16* buf, int n, int F, float eps,
                       u16 (*Ab)[32], const short (*nbr)[5], const float (*cond)[5],
                       int rowsPad, int k, int tid) {
  assume_lds(buf); assume_lds(Ab); assume_lds(nbr); assume_lds(cond);
  for (int r = tid; r < rowsPad; r += NTHREADS) {
    if (r < GB * n) {
      int g = r / n, i = r - g * n;
      const short* nb = nbr[g * N0 + i];
      for (int f = 0; f < F; ++f) {
        float v = (1.f + eps) * bf2f(buf[(g * N0 + i) * FMAX + f]);
        for (int kk = 0; kk < k; ++kk)
          v += bf2f(buf[(g * N0 + nb[kk]) * FMAX + f]);
        Ab[r][f] = f2bf(v);
      }
      for (int c = 0; c < 5; ++c) Ab[r][F + c] = f2bf(cond[g][c]);
      for (int c = F + 5; c < 32; ++c) Ab[r][c] = 0;
    } else {
      for (int c = 0; c < 32; ++c) Ab[r][c] = 0;
    }
  }
}

// ---------------- tsum_disc ----------------
// x = x + central_update(cnu0,x); x = x + central_update(cnu1,x);
// xd += disc(x.sum(nodes)).  Clobbers xt.
AINLINE void run_tsum(const KParams& kp, int T, int f, int n, u16* xt,
                      u16 (*Ab)[32], u16* Hw, float (*Hsm)[4], float (*Gv)[8],
                      float* xd, int tid, int lane, int wave)
{
  assume_lds(xt); assume_lds(Hsm); assume_lds(Gv); assume_lds(xd);
  int rowsPad = ((GB * n + 15) >> 4) << 4;
  for (int c = 0; c < 2; ++c) {
    int Cb = T + 12 * c;
    stage_plain(xt, n, f, Ab, rowsPad, tid);
    __syncthreads();
    FfnW We = ffnP(kp, Cb + 0, f, 4);                       // emb f->40->4, leaky out
    ffn_wmma(We, Ab, rowsPad, GB * n, Hw, true,
             nullptr, 0, nullptr, 0, &Hsm[0][0], 4, lane, wave);
    __syncthreads();
    if (tid < GB) {                                         // glob 4->40->5 on h.sum
      float s[4] = {0.f, 0.f, 0.f, 0.f};
      for (int i = 0; i < n; ++i)
        for (int j = 0; j < 4; ++j) s[j] += Hsm[tid * n + i][j];
      FfnW Wg = ffnP(kp, Cb + 4, 4, 5);
      ffn_scalar(s, Wg, true, &Gv[tid][0]);
    }
    __syncthreads();
    for (int r = tid; r < rowsPad; r += NTHREADS) {         // stage concat(h, g)
      if (r < GB * n) {
        int g = r / n;
        for (int j = 0; j < 4; ++j) Ab[r][j]     = f2bf(Hsm[r][j]);
        for (int j = 0; j < 5; ++j) Ab[r][4 + j] = f2bf(Gv[g][j]);
        for (int j = 9; j < 32; ++j) Ab[r][j] = 0;
      } else {
        for (int j = 0; j < 32; ++j) Ab[r][j] = 0;
      }
    }
    __syncthreads();
    FfnW Wo = ffnP(kp, Cb + 8, 9, f);                       // out 9->40->f, linear
    ffn_wmma(Wo, Ab, rowsPad, GB * n, Hw, false,
             xt, n, xt, /*resMode=2: x_new = 2x + out*/ 2, nullptr, 0, lane, wave);
    __syncthreads();
  }
  if (tid < GB) {                                           // disc f->40->1 on x.sum
    float s[FMAX];
    for (int j = 0; j < f; ++j) s[j] = 0.f;
    for (int i = 0; i < n; ++i)
      for (int j = 0; j < f; ++j) s[j] += bf2f(xt[(tid * N0 + i) * FMAX + j]);
    FfnW Wd; Wd.w1 = kp.p[T+26]; Wd.b1 = kp.p[T+24]; Wd.w2 = kp.p[T+27]; Wd.b2 = kp.p[T+25];
    Wd.g1 = nullptr; Wd.bt1 = nullptr; Wd.m1 = nullptr; Wd.v1 = nullptr;
    Wd.nin = f; Wd.nout = 1;
    float d;
    ffn_scalar(s, Wd, false, &d);
    xd[tid] += d;
  }
  __syncthreads();
}

// ---------------- one level ----------------
// knn -> gin0 -> gin1(+res) -> out-ffn -> lat -> tsum -> sag_pool
// input x in P, pooled output left in Q (P,Q clobbered).
AINLINE void run_level(const KParams& kp, int L, u16* P, u16* Q,
                       int n, int fin, int fout, int ratio, int latOff,
                       u16 (*Ab)[32], u16* Hw, float (*Hsm)[4], float (*cond)[5],
                       float (*Gv)[8], short (*nbr)[5], float* score, int (*perm)[8],
                       float* xd, float* outp, int gblk,
                       int tid, int lane, int wave)
{
  assume_lds(P); assume_lds(Q); assume_lds(score); assume_lds(perm); assume_lds(nbr);
  assume_glb(outp);
  int rowsPad = ((GB * n + 15) >> 4) << 4;
  int k = (n - 1 < 5) ? (n - 1) : 5;

  // --- knn on coords (features 0,1) ---
  for (int r = tid; r < GB * n; r += NTHREADS) {
    int g = r / n, i = r - g * n;
    float xi = bf2f(P[(g * N0 + i) * FMAX + 0]);
    float yi = bf2f(P[(g * N0 + i) * FMAX + 1]);
    float d[N0];
    for (int j = 0; j < n; ++j) {
      float dx = xi - bf2f(P[(g * N0 + j) * FMAX + 0]);
      float dy = yi - bf2f(P[(g * N0 + j) * FMAX + 1]);
      d[j] = (j == i) ? POS_INF : (dx * dx + dy * dy);
    }
    for (int kk = 0; kk < k; ++kk) {
      int am = 0; float bv = POS_INF;
      for (int j = 0; j < n; ++j) if (d[j] < bv) { bv = d[j]; am = j; }
      nbr[g * N0 + i][kk] = (short)am;
      d[am] = POS_INF;
    }
  }
  __syncthreads();

  // --- gin0: P(fin) -> Q(10), no residual ---
  float eps0 = kp.p[L + 28][0];
  stage_gin(P, n, fin, eps0, Ab, nbr, cond, rowsPad, k, tid);
  __syncthreads();
  { FfnW W = ffnBN(kp, L + 29, fin + 5, 10);
    ffn_wmma(W, Ab, rowsPad, GB * n, Hw, true, Q, n, nullptr, 0, nullptr, 0, lane, wave); }
  __syncthreads();

  // --- gin1: Q(10) -> Q(10), residual (10==10) ---
  float eps1 = kp.p[L + 37][0];
  stage_gin(Q, n, 10, eps1, Ab, nbr, cond, rowsPad, k, tid);
  __syncthreads();
  { FfnW W = ffnBN(kp, L + 38, 15, 10);
    ffn_wmma(W, Ab, rowsPad, GB * n, Hw, true, Q, n, Q, 1, nullptr, 0, lane, wave); }
  __syncthreads();

  // --- out ffn: Q(10) -> P(fout), bn, final-linear ---
  stage_plain(Q, n, 10, Ab, rowsPad, tid);
  __syncthreads();
  { FfnW W = ffnBN(kp, L + 46, 10, fout);
    ffn_wmma(W, Ab, rowsPad, GB * n, Hw, false, P, n, nullptr, 0, nullptr, 0, lane, wave); }
  __syncthreads();

  // --- lat: sum/max over nodes ---
  for (int i = tid; i < GB * fout; i += NTHREADS) {
    int g = i / fout, f = i - g * fout;
    float s = 0.f, m = NEG_INF;
    for (int j = 0; j < n; ++j) {
      float v = bf2f(P[(g * N0 + j) * FMAX + f]);
      s += v; m = fmaxf(m, v);
    }
    float* row = outp + BGRAPH + (size_t)(gblk + g) * 66;
    row[latOff + f]        = s;
    row[latOff + fout + f] = m;
  }
  // --- copy P -> Q as tsum's private working copy ---
  for (int i = tid; i < GB * n * fout; i += NTHREADS) {
    int g = i / (n * fout), rem = i - g * (n * fout);
    int j = rem / fout, f = rem - j * fout;
    Q[(g * N0 + j) * FMAX + f] = P[(g * N0 + j) * FMAX + f];
  }
  __syncthreads();
  run_tsum(kp, L + 0, fout, n, Q, Ab, Hw, Hsm, Gv, xd, tid, lane, wave);

  // --- sag_pool: linear score, top-ratio, gather P -> Q ---
  const float* pw = kp.p[L + 55];
  assume_glb(pw);
  float pb = kp.p[L + 54][0];
  for (int r = tid; r < GB * n; r += NTHREADS) {
    int g = r / n, i = r - g * n;
    float s = pb;
    for (int f = 0; f < fout; ++f) s += bf2f(P[(g * N0 + i) * FMAX + f]) * pw[f];
    score[g * N0 + i] = s;
  }
  __syncthreads();
  if (tid < GB) {
    for (int kk = 0; kk < ratio; ++kk) {
      int am = 0; float bv = NEG_INF;
      for (int j = 0; j < n; ++j) {
        float v = score[tid * N0 + j];
        if (v > bv) { bv = v; am = j; }
      }
      perm[tid][kk] = am;
      score[tid * N0 + am] = NEG_INF;
    }
  }
  __syncthreads();
  for (int i = tid; i < GB * ratio * fout; i += NTHREADS) {
    int g = i / (ratio * fout), rem = i - g * (ratio * fout);
    int j = rem / fout, f = rem - j * fout;
    Q[(g * N0 + j) * FMAX + f] = P[(g * N0 + perm[g][j]) * FMAX + f];
  }
  __syncthreads();
}

// ---------------- kernel ----------------
__global__ __launch_bounds__(NTHREADS, 1)
void gnn_fwd(KParams kp, float* __restrict__ outp)
{
  __shared__ u16   sP[GB * N0 * FMAX];     // ping feature buffer (bf16)
  __shared__ u16   sQ[GB * N0 * FMAX];     // pong
  __shared__ u16   sAb[RMAX][32];          // WMMA A staging
  __shared__ u16   sHb[NWAVES][16 * 64];   // per-wave hidden tiles
  __shared__ float sHsm[RMAX][4];
  __shared__ float sCond[GB][5];
  __shared__ float sGv[GB][8];
  __shared__ short sNbr[GB * N0][5];
  __shared__ float sScore[GB * N0];
  __shared__ int   sPerm[GB][8];
  __shared__ float sXd[GB];

  const int tid  = threadIdx.x;
  const int lane = tid & 31;
  const int wave = tid >> 5;
  const int gblk = blockIdx.x * GB;

  u16* Hw = sHb[wave];

  assume_glb(kp.x); assume_glb(kp.cond); assume_glb(outp);

  for (int i = tid; i < NWAVES * 16 * 64; i += NTHREADS) ((u16*)sHb)[i] = 0;
  if (tid < GB) sXd[tid] = 0.f;
  for (int i = tid; i < GB * 5; i += NTHREADS)
    ((float*)sCond)[i] = kp.cond[(size_t)gblk * 5 + i];
  for (int i = tid; i < GB * N0 * 3; i += NTHREADS) {
    int g = i / (N0 * 3), rem = i - g * (N0 * 3);
    int nn2 = rem / 3, f = rem - nn2 * 3;
    sP[(g * N0 + nn2) * FMAX + f] = f2bf(kp.x[((size_t)(gblk + g) * N0 + nn2) * 3 + f]);
  }
  __syncthreads();

  // lat of raw input x (3 feats): cols 0..2 sum, 3..5 max
  for (int i = tid; i < GB * 3; i += NTHREADS) {
    int g = i / 3, f = i - g * 3;
    float s = 0.f, m = NEG_INF;
    for (int j = 0; j < N0; ++j) {
      float v = bf2f(sP[(g * N0 + j) * FMAX + f]);
      s += v; m = fmaxf(m, v);
    }
    float* row = outp + BGRAPH + (size_t)(gblk + g) * 66;
    row[f] = s; row[3 + f] = m;
  }
  __syncthreads();

  run_level(kp, 28, sP, sQ, N0, 3, 12, 6, 6,
            sAb, Hw, sHsm, sCond, sGv, sNbr, sScore, sPerm, sXd,
            outp, gblk, tid, lane, wave);                                  // level0
  run_level(kp, 84, sQ, sP, 6, 12, 18, 1, 30,
            sAb, Hw, sHsm, sCond, sGv, sNbr, sScore, sPerm, sXd,
            outp, gblk, tid, lane, wave);                                  // level1
  run_tsum(kp, 0, 18, 1, sP, sAb, Hw, sHsm, sGv, sXd, tid, lane, wave);    // last_disc

  if (tid < GB) outp[gblk + tid] = sXd[tid];
}

// ---------------- host launch ----------------
extern "C" void kernel_launch(void* const* d_in, const int* in_sizes, int n_in,
                              void* d_out, int out_size, void* d_ws, size_t ws_size,
                              hipStream_t stream) {
  (void)out_size; (void)d_ws; (void)ws_size;
  KParams kp;
  int base;
  if (n_in >= 2 && in_sizes[0] == BGRAPH * 5) {
    // JAX pytree sorted-key order: condition, params..., x
    kp.cond = (const float*)d_in[0];
    kp.x    = (const float*)d_in[n_in - 1];
    base = 1;
  } else {
    // insertion order fallback: x, condition, params...
    kp.x    = (const float*)d_in[0];
    kp.cond = (const float*)d_in[1];
    base = 2;
  }
  for (int i = 0; i < 140; ++i) {
    int j = base + i;
    if (j >= n_in) j = n_in - 1;
    kp.p[i] = (const float*)d_in[j];
  }
  dim3 grid(BGRAPH / GB), block(NTHREADS);
  gnn_fwd<<<grid, block, 0, stream>>>(kp, (float*)d_out);
}
